// SeriesMLP_72748156060021
// MI455X (gfx1250) — matrix operand order repaired and checked by reading, not compiled, not run on hardware
//
#include <hip/hip_runtime.h>
#include <hip/hip_bf16.h>

// ---------------------------------------------------------------------------
// SeriesMLP for MI455X (gfx1250): batched per-series MLP via f16 WMMA +
// async global->LDS weight staging (CDNA5 ASYNCcnt path).
//   N=200000 points, D_IN=3, H=256, L=4 hidden layers, S=8 series.
// ---------------------------------------------------------------------------

typedef __attribute__((ext_vector_type(16))) _Float16 v16h;
typedef __attribute__((ext_vector_type(8)))  float    v8f;
typedef __attribute__((ext_vector_type(4)))  int      v4i;

typedef __attribute__((address_space(1))) v4i* gv4i_ptr;   // global int4*
typedef __attribute__((address_space(3))) v4i* lv4i_ptr;   // LDS int4*

#define NPTS   200000
#define HID    256
#define NSER   8
#define NLAY   4
#define MT     128          // points per block
#define AST    264          // act row stride in f16 (256+8 pad: 528B, 16B aligned)

#if defined(__has_builtin)
#if __has_builtin(__builtin_amdgcn_global_load_async_to_lds_b128)
#define HAVE_ASYNC_LDS 1
#endif
#endif
#ifndef HAVE_ASYNC_LDS
#define HAVE_ASYNC_LDS 0
#endif

union V16 { v16h v; uint4 q[2]; };

__device__ __forceinline__ float softplus_f(float x) {
    return fmaxf(x, 0.0f) + __logf(1.0f + __expf(-fabsf(x)));
}

// One 16-byte global->LDS transfer (async DMA if available).
__device__ __forceinline__ void stage16(const uint4* g, uint4* l) {
#if HAVE_ASYNC_LDS
    __builtin_amdgcn_global_load_async_to_lds_b128(
        (gv4i_ptr)(uintptr_t)g, (lv4i_ptr)l, 0, 0);
#else
    *l = *g;
#endif
}

__device__ __forceinline__ void stage_fence() {
#if HAVE_ASYNC_LDS
#if __has_builtin(__builtin_amdgcn_s_wait_asynccnt)
    __builtin_amdgcn_s_wait_asynccnt(0);
#else
    asm volatile("s_wait_asynccnt 0x0" ::: "memory");
#endif
#endif
}

// ---------------------------------------------------------------------------
// Pre-pass: convert Wh [S,L,H,H] fp32 -> f16, pre-swizzled into WMMA
// B-fragment order [sl][kt][nt][lane][v] (dwords): per-lane fragment is
// 32 contiguous bytes.  B order used here (contiguous K, not ISA table order): N = lane&15,
// K = kt*32 + (lane>>4)*16 + 2v (+1 in high half of the dword).
// ---------------------------------------------------------------------------
__global__ __launch_bounds__(256) void pack_wh_kernel(
        const float* __restrict__ Wh, unsigned int* __restrict__ wpk) {
    int idx  = blockIdx.x * 256 + threadIdx.x;     // 0 .. 1048575
    int v    = idx & 7;
    int lane = (idx >> 3) & 31;
    int nt   = (idx >> 8) & 15;
    int kt   = (idx >> 12) & 7;
    int sl   = idx >> 15;                          // s*4 + l
    int n    = nt * 16 + (lane & 15);
    int k    = kt * 32 + ((lane >> 4) << 4) + 2 * v;
    const float* p = Wh + ((size_t)sl * HID + k) * HID + n;
    union { _Float16 h[2]; unsigned int u; } pk;
    pk.h[0] = (_Float16)p[0];
    pk.h[1] = (_Float16)p[HID];
    wpk[idx] = pk.u;
}

// ---------------------------------------------------------------------------
// Fused MLP: one block = (128-point tile, series).  8 waves: wave w owns
// rows [(w>>1)*32,+32) x features [(w&1)*128,+128).  Per-ktile weights
// (16 KB) staged into a double-buffered LDS ring via async DMA, prefetched
// one ktile ahead; activations ping in place in a single padded LDS buffer.
// ---------------------------------------------------------------------------
__global__ __launch_bounds__(256) void series_mlp_kernel(
        const float* __restrict__ coords,
        const float* __restrict__ W0,
        const float* __restrict__ b0,
        const float* __restrict__ bh,
        const float* __restrict__ Wout,
        const float* __restrict__ bout,
        const unsigned int* __restrict__ wpk,
        float* __restrict__ out) {
    __shared__ _Float16 actS[MT * AST];     // 67,584 B
    __shared__ uint4    wbuf[2][1024];      // 2 x 16 KB weight ring (also scratch)

    const int tid  = threadIdx.x;
    const int lane = tid & 31;
    const int w    = tid >> 5;
    const int s    = blockIdx.y;
    const int m0   = blockIdx.x * MT;

    // ---------------- Layer 0: coords (K=3) -> actS ------------------------
    // Stage coords tile (384 floats) into LDS scratch, then feature-per-thread:
    // each thread loads its 4 weight scalars ONCE and sweeps the 128 rows.
    {
        float* cbuf = (float*)&wbuf[0][0];
        int g = m0 * 3 + tid;
        if (tid < 3 * MT)       cbuf[tid]       = (g < NPTS * 3) ? coords[g] : 0.f;
        if (tid + 256 < 3 * MT) cbuf[tid + 256] = (g + 256 < NPTS * 3) ? coords[g + 256] : 0.f;
    }
    __syncthreads();
    {
        const float* cbuf = (const float*)&wbuf[0][0];
        int n = tid;
        float wd0 = W0[(size_t)s * 3 * HID + n];
        float wd1 = W0[(size_t)s * 3 * HID + HID + n];
        float wd2 = W0[(size_t)s * 3 * HID + 2 * HID + n];
        float bb  = b0[s * HID + n];
        #pragma unroll 4
        for (int m = 0; m < MT; m++) {
            float x = bb + cbuf[m * 3] * wd0 + cbuf[m * 3 + 1] * wd1
                         + cbuf[m * 3 + 2] * wd2;
            actS[m * AST + n] = (_Float16)softplus_f(x);
        }
    }
    __syncthreads();

    // Fragment addressing constants (ISA 7.12.2 layouts, wave32)
    const int halfL = lane >> 4;     // K-half (A/B), M-half (C)
    const int l15   = lane & 15;     // M row (A) / N col (B,C)
    const int wm    = w >> 1;        // rows [wm*32, +32)
    const int wn    = w & 1;         // features [wn*128, +128)

    const uint4* wpkQ = (const uint4*)wpk;

    v8f acc[2][8];

    for (int l = 0; l < NLAY; l++) {
        const uint4* wsrcL = wpkQ + (size_t)((s * NLAY + l) * 8) * 1024;

        // stage kt=0 (16 KB: 4 x b128 per thread, async)
        {
            const uint4* p = wsrcL + tid * 4;
            uint4* d = &wbuf[0][tid * 4];
            stage16(p + 0, d + 0); stage16(p + 1, d + 1);
            stage16(p + 2, d + 2); stage16(p + 3, d + 3);
        }
        #pragma unroll
        for (int mi = 0; mi < 2; mi++)
            #pragma unroll
            for (int nt = 0; nt < 8; nt++)
                acc[mi][nt] = v8f{0.f, 0.f, 0.f, 0.f, 0.f, 0.f, 0.f, 0.f};
        stage_fence();
        __syncthreads();

        for (int kt = 0; kt < 8; kt++) {
            // async-prefetch ktile kt+1 while this ktile computes
            if (kt < 7) {
                const uint4* p = wsrcL + (kt + 1) * 1024 + tid * 4;
                uint4* d = &wbuf[(kt + 1) & 1][tid * 4];
                stage16(p + 0, d + 0); stage16(p + 1, d + 1);
                stage16(p + 2, d + 2); stage16(p + 3, d + 3);
            }
            // A fragments: rows wm*32 + {0,16} + l15, K window kt*32
            V16 A0, A1;
            {
                const _Float16* ab0 = actS + (wm * 32 + l15) * AST + kt * 32 + halfL * 16;
                const _Float16* ab1 = ab0 + 16 * AST;
                A0.q[0] = *(const uint4*)(ab0);
                A0.q[1] = *(const uint4*)(ab0 + 8);
                A1.q[0] = *(const uint4*)(ab1);
                A1.q[1] = *(const uint4*)(ab1 + 8);
            }
            const uint4* wb = wbuf[kt & 1] + wn * 512;
            // software-pipelined B: load nt+1 while WMMAs consume nt
            V16 Bc, Bn;
            Bc.q[0] = wb[lane * 2];
            Bc.q[1] = wb[lane * 2 + 1];
            #pragma unroll
            for (int nt = 0; nt < 8; nt++) {
                if (nt < 7) {
                    Bn.q[0] = wb[(nt + 1) * 64 + lane * 2];
                    Bn.q[1] = wb[(nt + 1) * 64 + lane * 2 + 1];
                }
                acc[0][nt] = __builtin_amdgcn_wmma_f32_16x16x32_f16(
                    false, A0.v, false, Bc.v, (short)0, acc[0][nt], false, false);
                acc[1][nt] = __builtin_amdgcn_wmma_f32_16x16x32_f16(
                    false, A1.v, false, Bc.v, (short)0, acc[1][nt], false, false);
                Bc = Bn;
            }
            stage_fence();     // staged ktile kt+1 landed in LDS
            __syncthreads();   // ring slot reusable, staging visible
        }

        // Epilogue: bias + softplus -> actS (A reads finished at last barrier)
        {
            const float* bhp = bh + (size_t)(s * NLAY + l) * HID + wn * 128;
            #pragma unroll
            for (int nt = 0; nt < 8; nt++) {
                float bias = bhp[nt * 16 + l15];
                #pragma unroll
                for (int mi = 0; mi < 2; mi++) {
                    int mBase = wm * 32 + mi * 16 + halfL * 8;  // C: M = 8*half + r
                    #pragma unroll
                    for (int r = 0; r < 8; r++) {
                        float x = acc[mi][nt][r] + bias;
                        actS[(mBase + r) * AST + wn * 128 + nt * 16 + l15] =
                            (_Float16)softplus_f(x);
                    }
                }
            }
        }
        __syncthreads();
    }

    // ---------------- Output layer: 256 -> 1 dot ---------------------------
    {
        float* wobuf = (float*)&wbuf[0][0];        // 1 KB Wout stage
        wobuf[tid] = Wout[s * HID + tid];
    }
    __syncthreads();
    {
        const float*    wop = (const float*)&wbuf[0][0] + (tid & 1) * 128;
        const _Float16* ap  = actS + (tid >> 1) * AST + (tid & 1) * 128;
        float sum = 0.f;
        #pragma unroll 8
        for (int j = 0; j < 128; j++)
            sum += (float)ap[j] * wop[j];
        sum += __shfl_xor(sum, 1, 32);
        int m = tid >> 1;
        if ((tid & 1) == 0 && (m0 + m) < NPTS)
            out[(size_t)(m0 + m) * NSER + s] = sum + bout[s];
    }
}

// ---------------------------------------------------------------------------
extern "C" void kernel_launch(void* const* d_in, const int* in_sizes, int n_in,
                              void* d_out, int out_size, void* d_ws, size_t ws_size,
                              hipStream_t stream) {
    const float* coords = (const float*)d_in[0];
    const float* W0     = (const float*)d_in[1];
    const float* b0     = (const float*)d_in[2];
    const float* Wh     = (const float*)d_in[3];
    const float* bh     = (const float*)d_in[4];
    const float* Wout   = (const float*)d_in[5];
    const float* bout   = (const float*)d_in[6];
    float* out = (float*)d_out;
    unsigned int* wpk = (unsigned int*)d_ws;   // 4 MB packed f16 weights

    pack_wh_kernel<<<4096, 256, 0, stream>>>(Wh, wpk);

    dim3 grid((NPTS + MT - 1) / MT, NSER);
    series_mlp_kernel<<<grid, 256, 0, stream>>>(coords, W0, b0, bh, Wout, bout,
                                                wpk, out);

    (void)hipMemcpyAsync(out + (size_t)NPTS * NSER, coords,
                         (size_t)NPTS * 3 * sizeof(float),
                         hipMemcpyDeviceToDevice, stream);
}
